// RETAIN_16088947491371
// MI455X (gfx1250) — matrix operand order repaired and checked by reading, not compiled, not run on hardware
//
#include <hip/hip_runtime.h>
#include <hip/hip_bf16.h>
#include <math.h>

// ---------------- problem constants ----------------
#define BB 64
#define TT 128
#define VV 32
#define DD 128
#define CC 10000
#define OO 512
#define TD3 384   // 3*D

typedef __attribute__((ext_vector_type(16))) _Float16 v16h;
typedef __attribute__((ext_vector_type(8)))  float    v8f;
typedef __attribute__((ext_vector_type(4)))  float    v4f;

// ---------------- WMMA helper (wave32, gfx1250) ----------------
__device__ __forceinline__ v8f wmma16(v16h a, v16h b, v8f c) {
    // D = A(16x32 f16) * B(32x16 f16) + C(16x16 f32)
    return __builtin_amdgcn_wmma_f32_16x16x32_f16(
        false, a, false, b, (short)0, c, false, false);
}

// ---- packed fragment layouts: one contiguous v16h (32B) per lane per frag ----
// A-matrix 16x32 f16 layout (ISA 7.12.2): lane L holds row m=L%16, K-half lh=L/16,
// element e <-> k:  k = (e&7) | (lh<<3) | ((e&8)<<1)   =>   e = (k&7)|((k>>1)&8)
// Packed A storage: [mtile][ktile(4)][laneSlot(32)][e(16)], laneSlot = lh*16+m.
__device__ __forceinline__ int pa_addr(int mtile, int m, int d) {
    int kt = d >> 5, kl = d & 31;
    int lh = (kl >> 3) & 1;
    int e  = (kl & 7) | ((kl >> 1) & 8);
    return ((mtile * 4 + kt) * 32 + lh * 16 + m) * 16 + e;
}
// B-matrix 32x16 f16 layout: same k order as A: lane half (k>>3)&1, e = (k&7)|((k>>1)&8).
// Packed B storage: [ktile][ntile][laneSlot(32)][e(16)], laneSlot = ((k>>3)&1)*16 + n%16.
// Fragment load (both A and B): single 32-byte (v16h) load per lane.
__device__ __forceinline__ v16h load_frag(const _Float16* base, int tile) {
    int lane = threadIdx.x & 31;
    return *(const v16h*)(base + (tile * 32 + lane) * 16);
}

struct f8 { v4f lo, hi; };
__device__ __forceinline__ f8 ld8(const float* p) {
    f8 r; r.lo = *(const v4f*)p; r.hi = *(const v4f*)(p + 4); return r;
}
__device__ __forceinline__ float get8(const f8& r, int i) {
    return i < 4 ? r.lo[i] : r.hi[i - 4];
}
__device__ __forceinline__ float sigmoidf_(float x) {
    return 1.0f / (1.0f + __expf(-x));
}

// ---------------- 1. f32 weight [128 x N] -> packed-B f16 ----------------
__global__ void k_cvt_b(const float* __restrict__ src, _Float16* __restrict__ dst, int N) {
    int i = blockIdx.x * blockDim.x + threadIdx.x;
    if (i >= DD * N) return;
    int k = i / N, n = i % N;
    int kt = k >> 5, kl = k & 31;
    int slot = (((kl >> 3) & 1) << 4) + (n & 15);
    int tile = kt * (N >> 4) + (n >> 4);
    dst[(tile * 32 + slot) * 16 + ((kl & 7) | ((kl >> 1) & 8))] = (_Float16)src[i];
}

// ---------------- 2. embedding gather-sum: x[b,t,d] (row-major f32) ----------------
__global__ void k_embed(const int* __restrict__ codes, const float* __restrict__ emb,
                        float* __restrict__ x) {
    int bt = blockIdx.x, d = threadIdx.x;
    const int* c = codes + bt * VV;
    float s = 0.f;
#pragma unroll 4
    for (int v = 0; v < VV; ++v) s += emb[c[v] * DD + d];
    x[bt * DD + d] = s;
}

// ---------------- 3. lengths[b] ----------------
__global__ void k_lengths(const int* __restrict__ codes, int* __restrict__ lengths) {
    int b = threadIdx.x;
    if (b >= BB) return;
    int len = 0;
    for (int t = 0; t < TT; ++t) len += (codes[(b * TT + t) * VV] != 0) ? 1 : 0;
    lengths[b] = len;
}

// ---------------- 4. masked reverse -> rx (packed-A f16) ----------------
__global__ void k_reverse(const float* __restrict__ x, const int* __restrict__ lengths,
                          _Float16* __restrict__ rx) {
    int bt = blockIdx.x;
    int b = bt >> 7, j = bt & 127;
    int d = threadIdx.x;
    int src = lengths[b] - 1 - j;
    float v = (src >= 0) ? x[(b * TT + src) * DD + d] : 0.f;
    rx[pa_addr(bt >> 4, bt & 15, d)] = (_Float16)v;
}

// ---------------- 5. xp = rx @ kernel + bias[0], output in GRU-friendly order ----
// xp layout: [t][btile(4)][col(384)][m(16)]  (same total size as [B,T,384])
// grid (512, 2), block 256 (8 waves); wave w owns N-tiles 3w..3w+2 of 24.
__global__ void k_xp(const _Float16* __restrict__ rx,
                     const _Float16* __restrict__ wA, const _Float16* __restrict__ wB,
                     const float* __restrict__ a_bias, const float* __restrict__ b_bias,
                     float* __restrict__ xpA, float* __restrict__ xpB) {
    int mt = blockIdx.x;
    int g  = blockIdx.y;
    const _Float16* W  = g ? wB : wA;
    const float*   bi  = g ? b_bias : a_bias;   // bias row 0
    float*         out = g ? xpB : xpA;
    int wave = threadIdx.x >> 5, lane = threadIdx.x & 31;
    int lh = lane >> 4, nl = lane & 15;
    int r0 = mt * 16;
    int b = r0 >> 7, t0 = r0 & 127;
    int btile = b >> 4, mb = b & 15;
    v16h a[4];
#pragma unroll
    for (int kt = 0; kt < 4; ++kt) a[kt] = load_frag(rx, mt * 4 + kt);
#pragma unroll
    for (int nt = 0; nt < 3; ++nt) {
        int ntile = wave * 3 + nt;
        v8f acc = {};
#pragma unroll
        for (int kt = 0; kt < 4; ++kt)
            acc = wmma16(a[kt], load_frag(W, kt * 24 + ntile), acc);
        int n = ntile * 16 + nl;
        float bv = bi[n];
#pragma unroll
        for (int vr = 0; vr < 8; ++vr) {
            int m = vr + 8 * lh;                 // row in tile -> t = t0+m
            out[(((size_t)(t0 + m) * 4 + btile) * TD3 + n) * 16 + mb] = acc[vr] + bv;
        }
    }
}

// ---------------- 6. GRU recurrence (persistent; rec + h16 in LDS) ----------------
// grid (4,2): x = 16-row batch tile, y = GRU id. block 256 = 8 waves.
// LDS: packed-B rec f16 (96KB) | packed-A h16 (4KB)  => 102400 B dynamic.
__global__ void k_gru(const _Float16* __restrict__ recA, const _Float16* __restrict__ recB,
                      const float* __restrict__ a_bias, const float* __restrict__ b_bias,
                      const float* __restrict__ xpA, const float* __restrict__ xpB,
                      const int* __restrict__ lengths,
                      float* __restrict__ ya32, _Float16* __restrict__ yb16) {
    extern __shared__ char smem[];
    _Float16* s_rec = (_Float16*)smem;                  // [4*24*32*16] packed-B
    _Float16* s_h16 = (_Float16*)(smem + DD * TD3 * 2); // [4*32*16]    packed-A

    int g  = blockIdx.y;
    int bx = blockIdx.x;
    const _Float16* rec   = g ? recB : recA;
    const float*    bias1 = (g ? b_bias : a_bias) + TD3;   // recurrent bias row
    const float*    xp    = g ? xpB : xpA;
    int tid = threadIdx.x, wave = tid >> 5, lane = tid & 31;
    int lh = lane >> 4, dl = lane & 15;
    int b0 = bx * 16;

    { // linear vector copy of packed rec into LDS
        const v4f* s = (const v4f*)rec;
        v4f* d = (v4f*)s_rec;
        for (int i = tid; i < (DD * TD3 * 2) / 16; i += 256) d[i] = s[i];
    }
    for (int i = tid; i < 4 * 32 * 16; i += 256) s_h16[i] = (_Float16)0.f;
    __syncthreads();

    int dcol = wave * 16 + dl;                  // this lane's feature dim
    float brz = bias1[dcol], brr = bias1[DD + dcol], brh = bias1[2 * DD + dcol];
    int lenr[8];
    float hreg[8];
#pragma unroll
    for (int vr = 0; vr < 8; ++vr) { lenr[vr] = lengths[b0 + vr + 8 * lh]; hreg[vr] = 0.f; }
    int h16a = pa_addr(0, 0, dcol);             // packed-A slot for (m=0, dcol); +16/m

    for (int t = 0; t < TT; ++t) {
        // prefetch next step's xp block
        int tn = (t + 1 < TT) ? (t + 1) : t;
        __builtin_prefetch(xp + (((size_t)tn * 4 + bx) * TD3 + dcol) * 16, 0, 1);

        v16h a[4];
#pragma unroll
        for (int kt = 0; kt < 4; ++kt) a[kt] = load_frag(s_h16, kt);
        __syncthreads();   // all reads of h done before any wave rewrites it

        v8f az = {}, ar = {}, ah = {};
#pragma unroll
        for (int kt = 0; kt < 4; ++kt) {
            az = wmma16(a[kt], load_frag(s_rec, kt * 24 + wave),      az);
            ar = wmma16(a[kt], load_frag(s_rec, kt * 24 + wave + 8),  ar);
            ah = wmma16(a[kt], load_frag(s_rec, kt * 24 + wave + 16), ah);
        }

        const float* xb = xp + (((size_t)t * 4 + bx) * TD3 + dcol) * 16 + 8 * lh;
        f8 xz8 = ld8(xb);
        f8 xr8 = ld8(xb + (size_t)DD * 16);
        f8 xh8 = ld8(xb + (size_t)2 * DD * 16);

#pragma unroll
        for (int vr = 0; vr < 8; ++vr) {
            int m = vr + 8 * lh;
            float z  = sigmoidf_(get8(xz8, vr) + az[vr] + brz);
            float r  = sigmoidf_(get8(xr8, vr) + ar[vr] + brr);
            float hc = tanhf(get8(xh8, vr) + r * (ah[vr] + brh));
            float hold = hreg[vr];
            float hn   = z * hold + (1.f - z) * hc;
            float hout = (t < lenr[vr]) ? hn : hold;   // masked carry
            hreg[vr] = hout;
            s_h16[h16a + m * 16] = (_Float16)hout;
            size_t row = (size_t)(b0 + m) * TT + t;
            if (g == 0) ya32[row * DD + dcol] = hout;                     // for alpha
            else        yb16[pa_addr((int)(row >> 4), (int)(row & 15), dcol)]
                            = (_Float16)hout;                              // for beta
        }
        __syncthreads();   // h16 update visible before next step's fragment loads
    }
}

// ---------------- 7. alpha logits + softmax over T ----------------
__global__ void k_alpha(const float* __restrict__ ya32, const float* __restrict__ alpha_w,
                        const float* __restrict__ alpha_b, float* __restrict__ alpha) {
    __shared__ float red[TT];
    int b = blockIdx.x, t = threadIdx.x;
    const v4f* gr = (const v4f*)(ya32 + ((size_t)b * TT + t) * DD);
    const v4f* aw = (const v4f*)alpha_w;
    float s = 0.f;
#pragma unroll 4
    for (int d = 0; d < DD / 4; ++d) {
        v4f g4 = gr[d], w4 = aw[d];
        s += g4[0]*w4[0] + g4[1]*w4[1] + g4[2]*w4[2] + g4[3]*w4[3];
    }
    s += alpha_b[0];
    red[t] = s; __syncthreads();
    for (int st = TT / 2; st > 0; st >>= 1) {
        if (t < st) red[t] = fmaxf(red[t], red[t + st]);
        __syncthreads();
    }
    float mx = red[0]; __syncthreads();
    float e = __expf(s - mx);
    red[t] = e; __syncthreads();
    for (int st = TT / 2; st > 0; st >>= 1) {
        if (t < st) red[t] += red[t + st];
        __syncthreads();
    }
    alpha[b * TT + t] = e / red[0];
}

// ---------------- 8. beta = tanh(h@beta_w+b); wbx = alpha*beta*x ----------------
// grid 512, block 256; wave w owns N-tile w (N=128).
__global__ void k_beta(const _Float16* __restrict__ yb16, const _Float16* __restrict__ betaW,
                       const float* __restrict__ beta_b, const float* __restrict__ alpha,
                       const float* __restrict__ x, float* __restrict__ wbx) {
    int mt = blockIdx.x;
    int wave = threadIdx.x >> 5, lane = threadIdx.x & 31;
    int lh = lane >> 4, nl = lane & 15;
    int r0 = mt * 16;
    v16h a[4];
#pragma unroll
    for (int kt = 0; kt < 4; ++kt) a[kt] = load_frag(yb16, mt * 4 + kt);
    v8f acc = {};
#pragma unroll
    for (int kt = 0; kt < 4; ++kt)
        acc = wmma16(a[kt], load_frag(betaW, kt * 8 + wave), acc);
    int d = wave * 16 + nl;
    float bb = beta_b[d];
#pragma unroll
    for (int vr = 0; vr < 8; ++vr) {
        size_t row = r0 + vr + 8 * lh;
        float v = tanhf(acc[vr] + bb) * alpha[row] * x[row * DD + d];
        wbx[row * DD + d] = v;
    }
}

// ---------------- 9. c[b,d] = sum_t wbx  -> packed-A f16 ----------------
__global__ void k_csum(const float* __restrict__ wbx, _Float16* __restrict__ c16) {
    int b = blockIdx.x, d = threadIdx.x;
    float s = 0.f;
#pragma unroll 4
    for (int t = 0; t < TT; ++t) s += wbx[((size_t)b * TT + t) * DD + d];
    c16[pa_addr(b >> 4, b & 15, d)] = (_Float16)s;
}

// ---------------- 10. out = c @ cls_w + cls_b ----------------
// grid (4,4), block 256; wave w owns N-tile ng*8+w of 32.
__global__ void k_cls(const _Float16* __restrict__ c16, const _Float16* __restrict__ clsW,
                      const float* __restrict__ cls_b, float* __restrict__ out) {
    int mt = blockIdx.x, ng = blockIdx.y;
    int wave = threadIdx.x >> 5, lane = threadIdx.x & 31;
    int lh = lane >> 4, nl = lane & 15;
    int r0 = mt * 16;
    v16h a[4];
#pragma unroll
    for (int kt = 0; kt < 4; ++kt) a[kt] = load_frag(c16, mt * 4 + kt);
    int ntile = ng * 8 + wave;
    v8f acc = {};
#pragma unroll
    for (int kt = 0; kt < 4; ++kt)
        acc = wmma16(a[kt], load_frag(clsW, kt * 32 + ntile), acc);
    int n = ntile * 16 + nl;
    float bv = cls_b[n];
#pragma unroll
    for (int vr = 0; vr < 8; ++vr)
        out[(size_t)(r0 + vr + 8 * lh) * OO + n] = acc[vr] + bv;
}

// ---------------- host launch ----------------
extern "C" void kernel_launch(void* const* d_in, const int* in_sizes, int n_in,
                              void* d_out, int out_size, void* d_ws, size_t ws_size,
                              hipStream_t stream) {
    const int*   codes   = (const int*)d_in[0];
    const float* emb     = (const float*)d_in[1];
    const float* a_kern  = (const float*)d_in[2];
    const float* a_rec   = (const float*)d_in[3];
    const float* a_bias  = (const float*)d_in[4];
    const float* b_kern  = (const float*)d_in[5];
    const float* b_rec   = (const float*)d_in[6];
    const float* b_bias  = (const float*)d_in[7];
    const float* alpha_w = (const float*)d_in[8];
    const float* alpha_b = (const float*)d_in[9];
    const float* beta_w  = (const float*)d_in[10];
    const float* beta_b  = (const float*)d_in[11];
    const float* cls_w   = (const float*)d_in[12];
    const float* cls_b   = (const float*)d_in[13];
    float* out = (float*)d_out;

    char* base = (char*)d_ws;
    size_t off = 0;
    auto take = [&](size_t bytes) -> char* {
        char* p = base + off;
        off = (off + bytes + 255) & ~(size_t)255;
        return p;
    };
    float*     x32   = (float*)    take((size_t)BB * TT * DD * 4);
    _Float16*  rx16  = (_Float16*) take((size_t)BB * TT * DD * 2);   // packed-A
    float*     xpA   = (float*)    take((size_t)BB * TT * TD3 * 4);  // [t][btile][col][m]
    float*     xpB   = (float*)    take((size_t)BB * TT * TD3 * 4);
    float*     ya32  = (float*)    take((size_t)BB * TT * DD * 4);
    _Float16*  yb16  = (_Float16*) take((size_t)BB * TT * DD * 2);   // packed-A
    float*     wbx   = (float*)    take((size_t)BB * TT * DD * 4);
    float*     alph  = (float*)    take((size_t)BB * TT * 4);
    _Float16*  c16   = (_Float16*) take((size_t)BB * DD * 2);        // packed-A
    int*       lens  = (int*)      take((size_t)BB * 4);
    _Float16*  kA16  = (_Float16*) take((size_t)DD * TD3 * 2);       // packed-B
    _Float16*  kB16  = (_Float16*) take((size_t)DD * TD3 * 2);
    _Float16*  rA16  = (_Float16*) take((size_t)DD * TD3 * 2);
    _Float16*  rB16  = (_Float16*) take((size_t)DD * TD3 * 2);
    _Float16*  bW16  = (_Float16*) take((size_t)DD * DD * 2);
    _Float16*  cW16  = (_Float16*) take((size_t)DD * OO * 2);

    // weight conversions to packed-B fragment order
    k_cvt_b<<<(DD * TD3 + 255) / 256, 256, 0, stream>>>(a_kern, kA16, TD3);
    k_cvt_b<<<(DD * TD3 + 255) / 256, 256, 0, stream>>>(b_kern, kB16, TD3);
    k_cvt_b<<<(DD * TD3 + 255) / 256, 256, 0, stream>>>(a_rec,  rA16, TD3);
    k_cvt_b<<<(DD * TD3 + 255) / 256, 256, 0, stream>>>(b_rec,  rB16, TD3);
    k_cvt_b<<<(DD * DD  + 255) / 256, 256, 0, stream>>>(beta_w, bW16, DD);
    k_cvt_b<<<(DD * OO  + 255) / 256, 256, 0, stream>>>(cls_w,  cW16, OO);

    k_embed  <<<BB * TT, DD, 0, stream>>>(codes, emb, x32);
    k_lengths<<<1, BB, 0, stream>>>(codes, lens);
    k_reverse<<<BB * TT, DD, 0, stream>>>(x32, lens, rx16);

    k_xp<<<dim3((BB * TT) / 16, 2), 256, 0, stream>>>(rx16, kA16, kB16,
                                                      a_bias, b_bias, xpA, xpB);

    size_t gru_lds = (size_t)DD * TD3 * 2 + 4 * 32 * 16 * 2;   // 96KB + 4KB
    k_gru<<<dim3(BB / 16, 2), 256, gru_lds, stream>>>(rA16, rB16, a_bias, b_bias,
                                                      xpA, xpB, lens, ya32, yb16);

    k_alpha<<<BB, TT, 0, stream>>>(ya32, alpha_w, alpha_b, alph);
    k_beta <<<(BB * TT) / 16, 256, 0, stream>>>(yb16, bW16, beta_b, alph, x32, wbx);
    k_csum <<<BB, DD, 0, stream>>>(wbx, c16);
    k_cls  <<<dim3(BB / 16, OO / 128), 256, 0, stream>>>(c16, cW16, cls_b, out);
}
